// GRUODEBayes_2284922602267
// MI455X (gfx1250) — compile-verified, matrix operand order checked by reading, not run on hardware
//
#include <hip/hip_runtime.h>
#include <hip/hip_bf16.h>
#include <math.h>

#define S_T   64
#define B_T   128
#define H_T   512
#define IN_T  64
#define E_T   10000
#define EMB_T 128

typedef __attribute__((ext_vector_type(16))) __bf16 v16bf;
typedef __attribute__((ext_vector_type(8)))  float  v8f;
typedef __attribute__((ext_vector_type(4)))  unsigned int u32x4;

union Frag { v16bf v; u32x4 u[2]; };

__device__ __forceinline__ unsigned short f2bf(float f) {
  unsigned u = __float_as_uint(f);
  u += 0x7FFFu + ((u >> 16) & 1u);   // round-to-nearest-even bf16
  return (unsigned short)(u >> 16);
}
__device__ __forceinline__ float sigmoidf_(float x) {
  return 1.0f / (1.0f + __expf(-x));
}

// ---------------------------------------------------------------------------
// f32 -> bf16 conversion
// ---------------------------------------------------------------------------
__global__ void conv_bf16_kernel(const float* __restrict__ src,
                                 unsigned short* __restrict__ dst, int n) {
  int i = blockIdx.x * 256 + threadIdx.x;
  if (i < n) dst[i] = f2bf(src[i]);
}

// ---------------------------------------------------------------------------
// Core GEMM: C[M=128, N] = A[128, K](bf16 row-major) * W[N, K]^T (+bias)(+relu)
// One wave -> 16(M) x 64(N); 8 waves/block -> 128 x 64 tile.
//
// B tile (64 cols x 32 k, shared by all 8 waves) is staged in LDS via the
// CDNA5 async global->LDS DMA path (ASYNCcnt), double-buffered so the next
// tile's copy overlaps the current tile's WMMAs. Column stride padded to
// 40 elems (80 B): 16-byte aligned ds_load_b128 fragments, 16 distinct banks
// across the 16 fragment lanes.
//
// A fragments are direct per-wave global loads (rows are wave-private and
// K-contiguous, matching the CDNA5 16-bit operand swizzle:
//   lanes 0-15 : row = lane,    K-halves {0..7},{16..23}
//   lanes 16-31: row = lane-16, K-halves {8..15},{24..31}).
// ---------------------------------------------------------------------------
#define BPAD 40   // padded column stride in bf16 elements (80 bytes)

__global__ __launch_bounds__(256) void gemm_bf16_tn(
    const unsigned short* __restrict__ A, int lda,
    const unsigned short* __restrict__ W, int ldw,
    const float* __restrict__ bias,
    float* __restrict__ C, int ldc,
    int N, int K, int act)
{
  __shared__ __align__(128) unsigned short Bs[2][64 * BPAD];

  const int tid   = threadIdx.x;
  const int lane  = tid & 31;
  const int wave  = tid >> 5;
  const int mt    = wave << 4;           // M tile base (M = 128 fixed)
  const int nblk  = blockIdx.x << 6;     // N block base (64 cols)
  const int rsel  = lane & 15;
  const int khalf = (lane & 16) ? 8 : 0;

  // async-copy assignment: thread t -> col = t>>2 (0..63), 16B chunk = t&3
  const int ccol = tid >> 2;
  const int cchk = tid & 3;
  int cn = nblk + ccol;
  if (cn >= N) cn = N - 1;               // clamp loads; stores are guarded
  const unsigned short* gsrc = W + (size_t)cn * ldw + cchk * 8;
  unsigned ldsoff0 = (unsigned)(uintptr_t)(&Bs[0][ccol * BPAD + cchk * 8]);
  unsigned ldsoff1 = (unsigned)(uintptr_t)(&Bs[1][ccol * BPAD + cchk * 8]);

  const unsigned short* arow = A + (size_t)(mt + rsel) * lda + khalf;

  v8f acc[4];
#pragma unroll
  for (int c = 0; c < 4; ++c)
#pragma unroll
    for (int j = 0; j < 8; ++j) acc[c][j] = 0.0f;

  const int nk = K >> 5;
  // prologue: tile 0 -> buffer 0 (ASYNCcnt = 1)
  asm volatile("global_load_async_to_lds_b128 %0, %1, off"
               :: "v"(ldsoff0), "v"(gsrc) : "memory");

  for (int kidx = 0; kidx < nk; ++kidx) {
    const int k = kidx << 5;
    if (kidx + 1 < nk) {
      // issue next tile into the other buffer, then retire only the current
      // tile (in-order ASYNCcnt): copy of tile k+1 overlaps WMMA of tile k
      unsigned loff = ((kidx + 1) & 1) ? ldsoff1 : ldsoff0;
      asm volatile("global_load_async_to_lds_b128 %0, %1, off"
                   :: "v"(loff), "v"(gsrc + (size_t)(k + 32)) : "memory");
      asm volatile("s_wait_asynccnt 0x1" ::: "memory");
    } else {
      asm volatile("s_wait_asynccnt 0x0" ::: "memory");
    }
    __syncthreads();   // publish tile kidx to all 8 waves

    Frag a;
    a.u[0] = *(const u32x4*)(arow + k);
    a.u[1] = *(const u32x4*)(arow + k + 16);
    __builtin_prefetch((const void*)(arow + k + 64), 0, 3);

    const unsigned short* bs = &Bs[kidx & 1][0];
#pragma unroll
    for (int c = 0; c < 4; ++c) {
      Frag b;
      const unsigned short* bp = bs + ((c << 4) + rsel) * BPAD + khalf;
      b.u[0] = *(const u32x4*)(bp);
      b.u[1] = *(const u32x4*)(bp + 16);
      acc[c] = __builtin_amdgcn_wmma_f32_16x16x32_bf16(
          false, a.v, false, b.v, (short)0, acc[c], false, false);
    }
    __syncthreads();   // all waves done reading before this buffer is reused
  }

  // C/D layout: lane<16 -> n = lane, rows mt+0..7; lane>=16 -> rows mt+8..15
  const int mrow = mt + ((lane & 16) ? 8 : 0);
#pragma unroll
  for (int c = 0; c < 4; ++c) {
    int n = nblk + (c << 4) + rsel;
    if (n >= N) continue;
    float bv = bias ? bias[n] : 0.0f;
#pragma unroll
    for (int j = 0; j < 8; ++j) {
      float v = acc[c][j] + bv;
      if (act) v = fmaxf(v, 0.0f);
      C[(size_t)(mrow + j) * ldc + n] = v;
    }
  }
}

// ---------------------------------------------------------------------------
// ODE elementwise: z/r gates, r*h in bf16
// zrpre: [B, 2H] (cols 0..511 = z-pre, 512..1023 = r-pre)
// ---------------------------------------------------------------------------
__global__ void ode_zr_kernel(const float* __restrict__ zrpre,
                              const float* __restrict__ hin,
                              float* __restrict__ zbuf,
                              unsigned short* __restrict__ rh_b)
{
  int i = blockIdx.x * 256 + threadIdx.x;
  if (i >= B_T * H_T) return;
  int b = i >> 9, j = i & (H_T - 1);
  float z = sigmoidf_(zrpre[(size_t)b * (2 * H_T) + j]);
  float r = sigmoidf_(zrpre[(size_t)b * (2 * H_T) + H_T + j]);
  zbuf[i] = z;
  rh_b[i] = f2bf(r * hin[i]);
}

// RK4 stage combine; dt = 0.25 (1/ODE_STEP)
__global__ void ode_k_kernel(const float* __restrict__ htpre,
                             const float* __restrict__ zbuf,
                             const float* __restrict__ hin,
                             const float* __restrict__ base,
                             float* __restrict__ acc,
                             float* __restrict__ outf,
                             unsigned short* __restrict__ outb,
                             int stage)
{
  int i = blockIdx.x * 256 + threadIdx.x;
  if (i >= B_T * H_T) return;
  float k = (1.0f - zbuf[i]) * (tanhf(htpre[i]) - hin[i]);
  float o;
  if (stage == 0)      { acc[i] = k;               o = base[i] + 0.125f * k; }
  else if (stage == 1) { acc[i] += 2.0f * k;       o = base[i] + 0.125f * k; }
  else if (stage == 2) { acc[i] += 2.0f * k;       o = base[i] + 0.25f  * k; }
  else                 { float a = acc[i] + k;     o = base[i] + (0.25f / 6.0f) * a; }
  outf[i] = o;
  outb[i] = f2bf(o);
}

// ---------------------------------------------------------------------------
// GRU cell elementwise
// ---------------------------------------------------------------------------
__global__ void gru_init_kernel(const float* __restrict__ gi,   // [B,3H] (incl. bih)
                                const float* __restrict__ bhh,  // [3H]
                                float* __restrict__ hcur,
                                unsigned short* __restrict__ hcur_b)
{
  int i = blockIdx.x * 256 + threadIdx.x;
  if (i >= B_T * H_T) return;
  int b = i >> 9, j = i & (H_T - 1);
  const float* g = gi + (size_t)b * (3 * H_T);
  float r = sigmoidf_(g[j] + bhh[j]);
  float z = sigmoidf_(g[H_T + j] + bhh[H_T + j]);
  float n = tanhf(g[2 * H_T + j] + r * bhh[2 * H_T + j]);
  float h = (1.0f - z) * n;              // prev h == 0
  hcur[i] = h;
  hcur_b[i] = f2bf(h);
}

__global__ void gru_combine_kernel(const float* __restrict__ gi,   // [B,3H]
                                   const float* __restrict__ gh,   // [B,3H]
                                   const float* __restrict__ hint, // post-ODE h
                                   float* __restrict__ hcur,       // in: pre-step h, out: new h
                                   unsigned short* __restrict__ hcur_b,
                                   const float* __restrict__ src,  // full src [S,B,IN]
                                   const int* __restrict__ trg,
                                   int step)
{
  int i = blockIdx.x * 256 + threadIdx.x;
  if (i >= B_T * H_T) return;
  int b = i >> 9, j = i & (H_T - 1);
  const float* gx = gi + (size_t)b * (3 * H_T);
  const float* gg = gh + (size_t)b * (3 * H_T);
  float r = sigmoidf_(gx[j] + gg[j]);
  float z = sigmoidf_(gx[H_T + j] + gg[H_T + j]);
  float n = tanhf(gx[2 * H_T + j] + r * gg[2 * H_T + j]);
  float hobs = (1.0f - z) * n + z * hint[i];
  bool active = trg[b] > step;
  float xlast = src[((size_t)step * B_T + b) * IN_T + (IN_T - 1)];
  bool obs = (xlast != -1.0f) && active;
  float hn = obs ? hobs : (active ? hint[i] : hcur[i]);
  hcur[i] = hn;
  hcur_b[i] = f2bf(hn);
}

// ---------------------------------------------------------------------------
// muti_task pieces
// ---------------------------------------------------------------------------
__global__ void eid_reduce_kernel(const float* __restrict__ logits,  // [B,E]
                                  int* __restrict__ idx_out,
                                  float* __restrict__ logz_out)
{
  int b = blockIdx.x, tid = threadIdx.x;
  __shared__ float sm[256];
  __shared__ int   si[256];
  const float* row = logits + (size_t)b * E_T;
  float m = -3.402823e38f; int mi = 0;
  for (int e = tid; e < E_T; e += 256) {
    float v = row[e];
    if (v > m) { m = v; mi = e; }
  }
  sm[tid] = m; si[tid] = mi;
  __syncthreads();
  for (int s = 128; s > 0; s >>= 1) {
    if (tid < s) {
      if (sm[tid + s] > sm[tid] ||
          (sm[tid + s] == sm[tid] && si[tid + s] < si[tid])) {
        sm[tid] = sm[tid + s]; si[tid] = si[tid + s];
      }
    }
    __syncthreads();
  }
  float rowmax = sm[0];
  int   rowidx = si[0];
  __syncthreads();
  float s = 0.0f;
  for (int e = tid; e < E_T; e += 256) s += __expf(row[e] - rowmax);
  sm[tid] = s;
  __syncthreads();
  for (int st = 128; st > 0; st >>= 1) {
    if (tid < st) sm[tid] += sm[tid + st];
    __syncthreads();
  }
  if (tid == 0) { idx_out[b] = rowidx; logz_out[b] = rowmax + __logf(sm[0]); }
}

__global__ void eid_write_kernel(const float* __restrict__ logits,
                                 const float* __restrict__ logz,
                                 const int* __restrict__ trg,
                                 int step, float* __restrict__ out)
{
  int t = blockIdx.x * 256 + threadIdx.x;      // B*E = 1,280,000
  int b = t / E_T;
  bool active = trg[b] > step;
  out[(size_t)step * B_T * E_T + t] = active ? (logits[t] - logz[b]) : 0.0f;
}

__global__ void concat_kernel(const float* __restrict__ emb,
                              const int* __restrict__ idx,
                              const unsigned short* __restrict__ h_b,
                              unsigned short* __restrict__ cat_b)
{
  const int W = EMB_T + H_T;                    // 640
  int i = blockIdx.x * 256 + threadIdx.x;
  if (i >= B_T * W) return;
  int b = i / W, c = i % W;
  unsigned short v;
  if (c < EMB_T) v = f2bf(emb[(size_t)idx[b] * EMB_T + c]);
  else           v = h_b[(size_t)b * H_T + (c - EMB_T)];
  cat_b[i] = v;
}

__global__ void rate_kernel(const float* __restrict__ r1,   // [B,H], post-relu
                            const float* __restrict__ Wr2,  // [1,H]
                            const float* __restrict__ br2,
                            const int* __restrict__ trg,
                            int step, float* __restrict__ out_rate)
{
  int b = blockIdx.x, tid = threadIdx.x;
  __shared__ float sm[256];
  float s = 0.0f;
  for (int j = tid; j < H_T; j += 256) s += r1[(size_t)b * H_T + j] * Wr2[j];
  sm[tid] = s;
  __syncthreads();
  for (int st = 128; st > 0; st >>= 1) {
    if (tid < st) sm[tid] += sm[tid + st];
    __syncthreads();
  }
  if (tid == 0) {
    float rate = sigmoidf_(sm[0] + br2[0]);
    bool active = trg[b] > step;
    out_rate[(size_t)step * B_T + b] = active ? rate : 0.0f;
  }
}

// ---------------------------------------------------------------------------
// Host side
// ---------------------------------------------------------------------------
static inline void launch_gemm(hipStream_t s, const unsigned short* A, int lda,
                               const unsigned short* W, int ldw,
                               const float* bias, float* C, int ldc,
                               int N, int K, int act) {
  dim3 grid((N + 63) / 64, 1, 1);
  gemm_bf16_tn<<<grid, 256, 0, s>>>(A, lda, W, ldw, bias, C, ldc, N, K, act);
}
static inline void launch_conv(hipStream_t s, const float* src,
                               unsigned short* dst, int n) {
  conv_bf16_kernel<<<(n + 255) / 256, 256, 0, s>>>(src, dst, n);
}

extern "C" void kernel_launch(void* const* d_in, const int* in_sizes, int n_in,
                              void* d_out, int out_size, void* d_ws, size_t ws_size,
                              hipStream_t stream) {
  (void)in_sizes; (void)n_in; (void)out_size; (void)ws_size;

  const float* src  = (const float*)d_in[0];
  const int*   trg  = (const int*)d_in[1];
  const float* Wih  = (const float*)d_in[2];
  const float* Whh  = (const float*)d_in[3];
  const float* bih  = (const float*)d_in[4];
  const float* bhh  = (const float*)d_in[5];
  const float* Whz  = (const float*)d_in[6];
  const float* Whr  = (const float*)d_in[7];
  const float* Who  = (const float*)d_in[8];
  const float* emb  = (const float*)d_in[9];
  const float* Weid = (const float*)d_in[10];
  const float* beid = (const float*)d_in[11];
  const float* Wr1  = (const float*)d_in[12];
  const float* br1  = (const float*)d_in[13];
  const float* Wr2  = (const float*)d_in[14];
  const float* br2  = (const float*)d_in[15];

  float* out_eid  = (float*)d_out;                              // [S,B,E]
  float* out_rate = (float*)d_out + (size_t)S_T * B_T * E_T;    // [S,B]

  // workspace bump allocator (256B aligned)
  char* p = (char*)d_ws;
  auto alloc = [&](size_t bytes) -> void* {
    char* r = p; p += (bytes + 255) & ~(size_t)255; return (void*)r;
  };
  unsigned short* Wih_b  = (unsigned short*)alloc((size_t)3 * H_T * IN_T * 2);
  unsigned short* Whh_b  = (unsigned short*)alloc((size_t)3 * H_T * H_T * 2);
  unsigned short* Whzr_b = (unsigned short*)alloc((size_t)2 * H_T * H_T * 2);
  unsigned short* Who_b  = (unsigned short*)alloc((size_t)H_T * H_T * 2);
  unsigned short* Weid_b = (unsigned short*)alloc((size_t)E_T * H_T * 2);
  unsigned short* Wr1_b  = (unsigned short*)alloc((size_t)H_T * (H_T + EMB_T) * 2);
  unsigned short* src_b  = (unsigned short*)alloc((size_t)S_T * B_T * IN_T * 2);
  unsigned short* hcur_b = (unsigned short*)alloc((size_t)B_T * H_T * 2);
  unsigned short* hint_b = (unsigned short*)alloc((size_t)B_T * H_T * 2);
  unsigned short* hin_b  = (unsigned short*)alloc((size_t)B_T * H_T * 2);
  unsigned short* rh_b   = (unsigned short*)alloc((size_t)B_T * H_T * 2);
  unsigned short* cat_b  = (unsigned short*)alloc((size_t)B_T * (EMB_T + H_T) * 2);

  float* hcur  = (float*)alloc((size_t)B_T * H_T * 4);
  float* hint  = (float*)alloc((size_t)B_T * H_T * 4);
  float* hin_f = (float*)alloc((size_t)B_T * H_T * 4);
  float* acc   = (float*)alloc((size_t)B_T * H_T * 4);
  float* zbuf  = (float*)alloc((size_t)B_T * H_T * 4);
  float* zrpre = (float*)alloc((size_t)B_T * 2 * H_T * 4);
  float* htpre = (float*)alloc((size_t)B_T * H_T * 4);
  float* gi    = (float*)alloc((size_t)B_T * 3 * H_T * 4);
  float* gh    = (float*)alloc((size_t)B_T * 3 * H_T * 4);
  float* logits = (float*)alloc((size_t)B_T * E_T * 4);
  float* logz  = (float*)alloc((size_t)B_T * 4);
  int*   idx   = (int*)alloc((size_t)B_T * 4);
  float* r1    = (float*)alloc((size_t)B_T * H_T * 4);

  // ---- weight / input conversion to bf16 (deterministic each call) ----
  launch_conv(stream, Wih,  Wih_b,  3 * H_T * IN_T);
  launch_conv(stream, Whh,  Whh_b,  3 * H_T * H_T);
  launch_conv(stream, Whz,  Whzr_b, H_T * H_T);                 // z rows first
  launch_conv(stream, Whr,  Whzr_b + (size_t)H_T * H_T, H_T * H_T);
  launch_conv(stream, Who,  Who_b,  H_T * H_T);
  launch_conv(stream, Weid, Weid_b, E_T * H_T);
  launch_conv(stream, Wr1,  Wr1_b,  H_T * (H_T + EMB_T));
  launch_conv(stream, src,  src_b,  S_T * B_T * IN_T);

  // row 0 of both outputs is all zeros in the reference
  hipMemsetAsync(out_eid, 0, (size_t)B_T * E_T * sizeof(float), stream);
  hipMemsetAsync(out_rate, 0, (size_t)B_T * sizeof(float), stream);

  const int NBH = (B_T * H_T + 255) / 256;      // 256 blocks

  // ---- h0 = GRU(src[0], 0) ----
  launch_gemm(stream, src_b, IN_T, Wih_b, IN_T, bih, gi, 3 * H_T,
              3 * H_T, IN_T, 0);
  gru_init_kernel<<<NBH, 256, 0, stream>>>(gi, bhh, hcur, hcur_b);

  auto muti_task = [&](int step) {
    launch_gemm(stream, hcur_b, H_T, Weid_b, H_T, beid, logits, E_T,
                E_T, H_T, 0);
    eid_reduce_kernel<<<B_T, 256, 0, stream>>>(logits, idx, logz);
    eid_write_kernel<<<(B_T * E_T) / 256, 256, 0, stream>>>(
        logits, logz, trg, step, out_eid);
    concat_kernel<<<(B_T * (EMB_T + H_T) + 255) / 256, 256, 0, stream>>>(
        emb, idx, hcur_b, cat_b);
    launch_gemm(stream, cat_b, EMB_T + H_T, Wr1_b, EMB_T + H_T, br1, r1, H_T,
                H_T, EMB_T + H_T, 1 /*relu*/);
    rate_kernel<<<B_T, 256, 0, stream>>>(r1, Wr2, br2, trg, step, out_rate);
  };

  for (int i = 1; i < S_T; ++i) {
    // row i uses the pre-ODE hidden state; row S-1 is overwritten post-loop
    if (i < S_T - 1) muti_task(i);

    // ---- RK4: 3 micro-steps (h_ode[-1] semantics => ODE_STEP-1 steps) ----
    for (int ms = 0; ms < 3; ++ms) {
      const float*          base_f = (ms == 0) ? hcur   : hint;
      const unsigned short* base_b = (ms == 0) ? hcur_b : hint_b;
      for (int stage = 0; stage < 4; ++stage) {
        const unsigned short* hb = (stage == 0) ? base_b : hin_b;
        const float*          hf = (stage == 0) ? base_f : hin_f;
        // z,r pre-activations in one fused GEMM (Whz ++ Whr), N = 2H
        launch_gemm(stream, hb, H_T, Whzr_b, H_T, nullptr, zrpre, 2 * H_T,
                    2 * H_T, H_T, 0);
        ode_zr_kernel<<<NBH, 256, 0, stream>>>(zrpre, hf, zbuf, rh_b);
        // h_tilde pre-activation: (r*h) @ Whh_ode^T
        launch_gemm(stream, rh_b, H_T, Who_b, H_T, nullptr, htpre, H_T,
                    H_T, H_T, 0);
        float*          outf = (stage == 3) ? hint   : hin_f;
        unsigned short* outb = (stage == 3) ? hint_b : hin_b;
        ode_k_kernel<<<NBH, 256, 0, stream>>>(htpre, zbuf, hf, base_f, acc,
                                              outf, outb, stage);
      }
    }

    // ---- GRU observation update ----
    launch_gemm(stream, src_b + (size_t)i * B_T * IN_T, IN_T, Wih_b, IN_T,
                bih, gi, 3 * H_T, 3 * H_T, IN_T, 0);
    launch_gemm(stream, hint_b, H_T, Whh_b, H_T, bhh, gh, 3 * H_T,
                3 * H_T, H_T, 0);
    gru_combine_kernel<<<NBH, 256, 0, stream>>>(gi, gh, hint, hcur, hcur_b,
                                                src, trg, i);
  }

  // final row S-1 from post-loop hidden state
  muti_task(S_T - 1);
}
